// TowersMLPForDigit1H_46840913330186
// MI455X (gfx1250) — compile-verified, matrix operand order checked by reading, not run on hardware
//
#include <hip/hip_runtime.h>
#include <hip/hip_bf16.h>
#include <math.h>

// ---------------------------------------------------------------------------
// TowersMLPForDigit1H on MI455X (gfx1250): all matmuls via v_wmma_f32_16x16x32_f16.
// One block = 4 waves (wave32) cooperating on 16 rows (= 48 tokens, 3 M-tiles).
// N-tiles are distributed across waves; B (weight) fragments are hoisted into
// registers and reused across the 3 M-tiles, whose WMMA chains are interleaved
// (independent accumulators) to cover the WMMA data hazards. x is staged into
// padded LDS so all fragment loads are aligned b128 with uniform control flow.
// ---------------------------------------------------------------------------

typedef _Float16 half_t;
typedef __attribute__((ext_vector_type(16))) _Float16 v16h;
typedef __attribute__((ext_vector_type(8)))  _Float16 v8h;
typedef __attribute__((ext_vector_type(8)))  float    v8f;
typedef __attribute__((ext_vector_type(4)))  float    v4f;

// f16 weight workspace layout (element offsets, all multiples of 8 -> 16B aligned)
#define OWEMB 0                      // [80][32]   block-diag digit embedding (K pad 30->32, N pad 72->80)
#define OTW1  (OWEMB + 80*32)        // [256][96]  tw1, K pad 72->96
#define OTW2  (OTW1  + 256*96)       // [256][256]
#define OTW3  (OTW2  + 256*256)      // [64][256]
#define OINPW (OTW3  + 64*256)       // [192][64]
#define OOUTW (OINPW + 192*64)       // [64][64]
#define OFFW1 (OOUTW + 64*64)        // [128][64]
#define OFFW2 (OFFW1 + 128*64)       // [64][128]
#define OHW1  (OFFW2 + 64*128)       // [160][192]
#define OHW2  (OHW1  + 160*192)      // [64][160]
#define OEND  (OHW2  + 64*160)       // 182784 halfs (even -> float region aligned)
// float region appended after the f16 weights:
//   posd_pad[3][80]: stage-1 bias padded to 80 cols (zeros beyond 72)

// LDS strides (in elements). All chosen so row stride is a multiple of 16 bytes.
#define SXA 40    // xa: [48][32] f16 staged input (K pad 30->32)
#define SA0 104   // act0: [48][96] f16 region (stage-1 out / stage-2 A, K padded to 96)
#define SH  264   // h1/h2: [48][256] f16
#define SCK 68    // chunks residual: [48][64] f32
#define SAF 72    // ln out / attn-o / ffn-in: [48][64] f16
#define SQ  200   // qkv: [48][192] f16
#define SG  136   // ffn hidden: [48][128] f16
#define SHH 168   // head hidden1: [16][160] f16
#define SH2 68    // head hidden2: [16][64] f32

#define BUF0_OFF 0
#define BUF1_OFF 9984          // 48*SA0*2 = 9984
#define BUF2_OFF (9984+25344)  // + 48*SH*2
#define SMEM_BYTES (9984 + 25344 + 25344)  // 60672 B < 64 KB

#define NWAVES 4
#define NTHREADS (NWAVES * 32)

__device__ __forceinline__ float silu_f(float x) {
    return x / (1.0f + __expf(-x));
}

// Load one 16-lane-striped f16 fragment (A-or-B style) from a row-major
// [rows][ldk] f16 buffer. `p` must point at (row, k0 + (lane>>4)*8).
// CDNA5 16-bit layout: first 8 halfs = K kb..kb+7, next 8 = K kb+16..kb+23.
__device__ __forceinline__ v16h frag_ld(const half_t* p) {
    v8h lo = *(const v8h*)(p);
    v8h hi = *(const v8h*)(p + 16);
    v16h f;
#pragma unroll
    for (int i = 0; i < 8; ++i) { f[i] = lo[i]; f[8 + i] = hi[i]; }
    return f;
}

__device__ __forceinline__ v8f wmma_f16(v16h a, v16h b, v8f c) {
    return __builtin_amdgcn_wmma_f32_16x16x32_f16(false, a, false, b, (short)0, c,
                                                  false, false);
}

// Hoist all K-step B fragments (16 cols of W[N][K]) into registers.
template <int KS>
__device__ __forceinline__ void load_bfrags(v16h (&b)[KS], const half_t* B,
                                            int ldb, int lane) {
    int rlo = lane & 15;
    int lh8 = (lane >> 4) * 8;
    const half_t* pb = B + rlo * ldb + lh8;
#pragma unroll
    for (int ks = 0; ks < KS; ++ks) b[ks] = frag_ld(pb + ks * 32);
}

// One 16x16 output tile: A[16][K] (LDS) x pre-loaded B fragments.
template <int KS>
__device__ __forceinline__ v8f gemm_tile_b(const half_t* A, int lda,
                                           const v16h (&b)[KS], int lane) {
    int rlo = lane & 15;
    int lh8 = (lane >> 4) * 8;
    const half_t* pa = A + rlo * lda + lh8;
    v8f acc = {};
#pragma unroll
    for (int ks = 0; ks < KS; ++ks)
        acc = wmma_f16(frag_ld(pa + ks * 32), b[ks], acc);
    return acc;
}

// Three M-tiles (rows 0..47 of A) against shared B fragments, with the three
// independent WMMA chains interleaved per K-step to cover WMMA data hazards.
template <int KS>
__device__ __forceinline__ void gemm_tile_b3(const half_t* A, int lda,
                                             const v16h (&b)[KS], int lane,
                                             v8f (&d)[3]) {
    int rlo = lane & 15;
    int lh8 = (lane >> 4) * 8;
    const half_t* pa = A + rlo * lda + lh8;
    v8f a0 = {}, a1 = {}, a2 = {};
#pragma unroll
    for (int ks = 0; ks < KS; ++ks) {
        v16h f0 = frag_ld(pa + ks * 32);
        v16h f1 = frag_ld(pa + 16 * lda + ks * 32);
        v16h f2 = frag_ld(pa + 32 * lda + ks * 32);
        a0 = wmma_f16(f0, b[ks], a0);
        a1 = wmma_f16(f1, b[ks], a1);
        a2 = wmma_f16(f2, b[ks], a2);
    }
    d[0] = a0; d[1] = a1; d[2] = a2;
}

// 16 contiguous halfs -> float[16] (two 16B DS loads).
__device__ __forceinline__ void ld16f(float* d, const half_t* p) {
    v8h a = *(const v8h*)(p);
    v8h b = *(const v8h*)(p + 8);
#pragma unroll
    for (int i = 0; i < 8; ++i) { d[i] = (float)a[i]; d[8 + i] = (float)b[i]; }
}

// LayerNorm over 48 tokens x 64 features, vectorized 16B LDS traffic.
__device__ __forceinline__ void ln48(const float* cksb, const float* g,
                                     const float* b, half_t* dst, int tid) {
    const v4f* g4 = (const v4f*)g;
    const v4f* b4 = (const v4f*)b;
    for (int t = tid; t < 48; t += NTHREADS) {
        const v4f* row4 = (const v4f*)(cksb + t * SCK);
        v4f rv[16];
        float s = 0.f, s2 = 0.f;
#pragma unroll
        for (int i = 0; i < 16; ++i) {
            v4f v = row4[i];
            rv[i] = v;
#pragma unroll
            for (int j = 0; j < 4; ++j) { s += v[j]; s2 += v[j] * v[j]; }
        }
        float mu  = s * (1.f / 64.f);
        float var = s2 * (1.f / 64.f) - mu * mu;
        float inv = rsqrtf(var + 1e-5f);
        half_t* dr = dst + t * SAF;
#pragma unroll
        for (int i = 0; i < 8; ++i) {
            v4f a = rv[2 * i], c = rv[2 * i + 1];
            v4f ga = g4[2 * i], gc = g4[2 * i + 1];
            v4f ba = b4[2 * i], bc = b4[2 * i + 1];
            v8h o;
#pragma unroll
            for (int j = 0; j < 4; ++j) {
                o[j]     = (half_t)((a[j] - mu) * inv * ga[j] + ba[j]);
                o[4 + j] = (half_t)((c[j] - mu) * inv * gc[j] + bc[j]);
            }
            *(v8h*)(dr + i * 8) = o;
        }
    }
}

// ---------------------------------------------------------------------------
// Prep: convert all weights to f16 into workspace (with K padding / blocking),
// plus the padded stage-1 bias table (f32).
// ---------------------------------------------------------------------------
__global__ __launch_bounds__(256) void prep_kernel(
    const float* __restrict__ w_emb, const float* __restrict__ pos_d,
    const float* __restrict__ tw1,   const float* __restrict__ tw2,
    const float* __restrict__ tw3,   const float* __restrict__ inpw,
    const float* __restrict__ outw,  const float* __restrict__ ffw1,
    const float* __restrict__ ffw2,  const float* __restrict__ hw1,
    const float* __restrict__ hw2,   half_t* __restrict__ ws) {
    int tid = blockIdx.x * blockDim.x + threadIdx.x;
    int stp = gridDim.x * blockDim.x;
    // Block-diagonal digit embedding, stored [col j][k]: B[k][j] = w_emb[j%24][k%10]
    for (int i = tid; i < 80 * 32; i += stp) {
        int j = i >> 5, k = i & 31;
        float v = 0.f;
        if (j < 72 && k < 30 && (j / 24) == (k / 10))
            v = w_emb[(j % 24) * 10 + (k % 10)];
        ws[OWEMB + i] = (half_t)v;
    }
    for (int i = tid; i < 256 * 96; i += stp) {
        int n = i / 96, k = i % 96;
        ws[OTW1 + i] = (half_t)(k < 72 ? tw1[n * 72 + k] : 0.f);
    }
    for (int i = tid; i < 256 * 256; i += stp) ws[OTW2 + i] = (half_t)tw2[i];
    for (int i = tid; i < 64 * 256;  i += stp) ws[OTW3 + i] = (half_t)tw3[i];
    for (int i = tid; i < 192 * 64;  i += stp) ws[OINPW + i] = (half_t)inpw[i];
    for (int i = tid; i < 64 * 64;   i += stp) ws[OOUTW + i] = (half_t)outw[i];
    for (int i = tid; i < 128 * 64;  i += stp) ws[OFFW1 + i] = (half_t)ffw1[i];
    for (int i = tid; i < 64 * 128;  i += stp) ws[OFFW2 + i] = (half_t)ffw2[i];
    for (int i = tid; i < 160 * 192; i += stp) ws[OHW1 + i] = (half_t)hw1[i];
    for (int i = tid; i < 64 * 160;  i += stp) ws[OHW2 + i] = (half_t)hw2[i];
    // Padded stage-1 bias table: posd_pad[mt][n] (n in [0,80))
    float* posd_pad = (float*)(ws + OEND);
    for (int i = tid; i < 3 * 80; i += stp) {
        int mt = i / 80, n = i % 80;
        posd_pad[i] = (n < 72) ? pos_d[(mt * 3 + n / 24) * 24 + (n % 24)] : 0.f;
    }
}

// ---------------------------------------------------------------------------
// Main fused kernel: 4 waves per block, 16 rows per block.
// ---------------------------------------------------------------------------
__global__ __launch_bounds__(NTHREADS) void towers_main(
    const float* __restrict__ x,
    const float* __restrict__ pos_c,
    const float* __restrict__ tb1,   const float* __restrict__ tb2,
    const float* __restrict__ tb3,
    const float* __restrict__ ln1g,  const float* __restrict__ ln1b,
    const float* __restrict__ inpb,  const float* __restrict__ outb,
    const float* __restrict__ ln2g,  const float* __restrict__ ln2b,
    const float* __restrict__ ffb1,  const float* __restrict__ ffb2,
    const float* __restrict__ hb1,   const float* __restrict__ hb2,
    const float* __restrict__ hw3,   const float* __restrict__ hb3,
    const half_t* __restrict__ ws,
    float* __restrict__ out) {
    __shared__ __align__(16) char smem[SMEM_BYTES];
    half_t* act0 = (half_t*)(smem + BUF0_OFF);  // also: af16 / o / f / hh1
    half_t* h1   = (half_t*)(smem + BUF1_OFF);
    half_t* xa   = (half_t*)(smem + BUF1_OFF);  // staged x (dead before h1 written)
    float*  cks  = (float*)(smem + BUF1_OFF);   // overlays h1 (after h1 dead)
    half_t* h2   = (half_t*)(smem + BUF2_OFF);  // also: qkv / g / hh2(f32)
    half_t* qkvb = h2;
    half_t* gbuf = h2;
    float*  hh2f = (float*)(smem + BUF2_OFF);
    half_t* af16 = act0;
    half_t* obuf = act0;
    half_t* fbuf = act0;
    half_t* hh1  = act0;

    const int tid  = threadIdx.x;
    const int lane = tid & 31;
    const int wv   = tid >> 5;          // wave id 0..3
    const int rlo  = lane & 15;
    const int lh8  = (lane >> 4) * 8;
    const int mo   = lh8;               // D-tile M offset for upper lanes
    const int row0 = blockIdx.x * 16;
    const float* posd_pad = (const float*)(ws + OEND);

    // ---- Stage 0: stage x into padded LDS: xa[t=mt*16+r][k] = x[r][30*mt+k], k<30
    for (int idx = tid; idx < 48 * 32; idx += NTHREADS) {
        int t = idx >> 5, k = idx & 31;
        int mt = t >> 4, r = t & 15;
        int kc = (k < 30) ? k : 0;      // keep load in-bounds, select after
        float v = x[(size_t)(row0 + r) * 90 + 30 * mt + kc];
        xa[t * SXA + k] = (half_t)((k < 30) ? v : 0.f);
    }
    __syncthreads();

    // ---- Stage 1: chunks0 = blockdiag(x) @ Wemb + posd_pad -> act0[48][<=80] f16
    for (int nt = wv; nt < 5; nt += NWAVES) {
        v16h bf[1];
        load_bfrags<1>(bf, ws + OWEMB + nt * 16 * 32, 32, lane);
        int n = nt * 16 + rlo;
        v8f d[3];
        gemm_tile_b3<1>(xa, SXA, bf, lane, d);
#pragma unroll
        for (int mt = 0; mt < 3; ++mt) {
            float bias = posd_pad[mt * 80 + n];
#pragma unroll
            for (int v = 0; v < 8; ++v)
                act0[(mt * 16 + mo + v) * SA0 + n] = (half_t)(d[mt][v] + bias);
        }
    }
    // zero K-padding cols 80..95 (stage-2 reads K up to 96)
    for (int idx = tid; idx < 48 * 16; idx += NTHREADS)
        act0[(idx >> 4) * SA0 + 80 + (idx & 15)] = (half_t)0.f;
    __syncthreads();

    // ---- Stage 2: h1 = silu(chunks0 @ tw1.T + tb1)   [48][256]
    for (int nt = wv; nt < 16; nt += NWAVES) {
        v16h bf[3];
        load_bfrags<3>(bf, ws + OTW1 + nt * 16 * 96, 96, lane);
        int n = nt * 16 + rlo;
        float bias = tb1[n];
        v8f d[3];
        gemm_tile_b3<3>(act0, SA0, bf, lane, d);
#pragma unroll
        for (int mt = 0; mt < 3; ++mt)
#pragma unroll
            for (int v = 0; v < 8; ++v)
                h1[(mt * 16 + mo + v) * SH + n] = (half_t)silu_f(d[mt][v] + bias);
    }
    __syncthreads();

    // ---- Stage 3: h2 = silu(h1 @ tw2.T + tb2)   [48][256]
    for (int nt = wv; nt < 16; nt += NWAVES) {
        v16h bf[8];
        load_bfrags<8>(bf, ws + OTW2 + nt * 16 * 256, 256, lane);
        int n = nt * 16 + rlo;
        float bias = tb2[n];
        v8f d[3];
        gemm_tile_b3<8>(h1, SH, bf, lane, d);
#pragma unroll
        for (int mt = 0; mt < 3; ++mt)
#pragma unroll
            for (int v = 0; v < 8; ++v)
                h2[(mt * 16 + mo + v) * SH + n] = (half_t)silu_f(d[mt][v] + bias);
    }
    __syncthreads();

    // ---- Stage 4: chunks = h2 @ tw3.T + tb3 + pos_c -> cks[48][64] f32 (overlays h1)
    for (int nt = wv; nt < 4; nt += NWAVES) {
        v16h bf[8];
        load_bfrags<8>(bf, ws + OTW3 + nt * 16 * 256, 256, lane);
        int n = nt * 16 + rlo;
        float b0 = tb3[n];
        v8f d[3];
        gemm_tile_b3<8>(h2, SH, bf, lane, d);
#pragma unroll
        for (int mt = 0; mt < 3; ++mt) {
            float bias = b0 + pos_c[mt * 64 + n];
#pragma unroll
            for (int v = 0; v < 8; ++v)
                cks[(mt * 16 + mo + v) * SCK + n] = d[mt][v] + bias;
        }
    }
    __syncthreads();

    // ---- LayerNorm 1 -> af16[48][64] f16 (overlays act0)
    ln48(cks, ln1g, ln1b, af16, tid);
    __syncthreads();

    // ---- Stage 5: qkv = a @ inpw.T + inpb -> qkvb[48][192] f16 (overlays h2)
    for (int nt = wv; nt < 12; nt += NWAVES) {
        v16h bf[2];
        load_bfrags<2>(bf, ws + OINPW + nt * 16 * 64, 64, lane);
        int n = nt * 16 + rlo;
        float bias = inpb[n];
        v8f d[3];
        gemm_tile_b3<2>(af16, SAF, bf, lane, d);
#pragma unroll
        for (int mt = 0; mt < 3; ++mt)
#pragma unroll
            for (int v = 0; v < 8; ++v)
                qkvb[(mt * 16 + mo + v) * SQ + n] = (half_t)(d[mt][v] + bias);
    }
    __syncthreads();

    // ---- Attention (3 tokens, 4 heads, d=16) in VALU -> obuf[48][64] f16
    if (tid < 64) {
        int r  = tid & 15;
        int hh = tid >> 4;
        const half_t* base = qkvb + r * SQ + 16 * hh;
        float sc[3][3];
#pragma unroll
        for (int s = 0; s < 3; ++s) {
            float qd[16];
            ld16f(qd, base + s * 16 * SQ);
#pragma unroll
            for (int t2 = 0; t2 < 3; ++t2) {
                float kd[16];
                ld16f(kd, base + t2 * 16 * SQ + 64);
                float acc = 0.f;
#pragma unroll
                for (int dd = 0; dd < 16; ++dd) acc += qd[dd] * kd[dd];
                sc[s][t2] = acc * 0.25f;
            }
        }
        float vv[3][16];
#pragma unroll
        for (int t2 = 0; t2 < 3; ++t2) ld16f(vv[t2], base + t2 * 16 * SQ + 128);
#pragma unroll
        for (int s = 0; s < 3; ++s) {
            float m  = fmaxf(sc[s][0], fmaxf(sc[s][1], sc[s][2]));
            float e0 = __expf(sc[s][0] - m);
            float e1 = __expf(sc[s][1] - m);
            float e2 = __expf(sc[s][2] - m);
            float inv = 1.f / (e0 + e1 + e2);
            float w0 = e0 * inv, w1 = e1 * inv, w2 = e2 * inv;
            v8h o0, o1;
#pragma unroll
            for (int j = 0; j < 8; ++j) {
                o0[j] = (half_t)(w0 * vv[0][j] + w1 * vv[1][j] + w2 * vv[2][j]);
                o1[j] = (half_t)(w0 * vv[0][8 + j] + w1 * vv[1][8 + j] + w2 * vv[2][8 + j]);
            }
            half_t* op = obuf + (s * 16 + r) * SAF + 16 * hh;
            *(v8h*)(op)     = o0;
            *(v8h*)(op + 8) = o1;
        }
    }
    __syncthreads();

    // ---- Stage 6: chunks += o @ outw.T + outb
    for (int nt = wv; nt < 4; nt += NWAVES) {
        v16h bf[2];
        load_bfrags<2>(bf, ws + OOUTW + nt * 16 * 64, 64, lane);
        int n = nt * 16 + rlo;
        float bias = outb[n];
        v8f d[3];
        gemm_tile_b3<2>(obuf, SAF, bf, lane, d);
#pragma unroll
        for (int mt = 0; mt < 3; ++mt)
#pragma unroll
            for (int v = 0; v < 8; ++v)
                cks[(mt * 16 + mo + v) * SCK + n] += d[mt][v] + bias;
    }
    __syncthreads();

    // ---- LayerNorm 2 -> fbuf[48][64] f16
    ln48(cks, ln2g, ln2b, fbuf, tid);
    __syncthreads();

    // ---- Stage 7: g = silu(f @ ffw1.T + ffb1) -> gbuf[48][128] f16 (overlays qkv)
    for (int nt = wv; nt < 8; nt += NWAVES) {
        v16h bf[2];
        load_bfrags<2>(bf, ws + OFFW1 + nt * 16 * 64, 64, lane);
        int n = nt * 16 + rlo;
        float bias = ffb1[n];
        v8f d[3];
        gemm_tile_b3<2>(fbuf, SAF, bf, lane, d);
#pragma unroll
        for (int mt = 0; mt < 3; ++mt)
#pragma unroll
            for (int v = 0; v < 8; ++v)
                gbuf[(mt * 16 + mo + v) * SG + n] = (half_t)silu_f(d[mt][v] + bias);
    }
    __syncthreads();

    // ---- Stage 8: chunks += g @ ffw2.T + ffb2
    for (int nt = wv; nt < 4; nt += NWAVES) {
        v16h bf[4];
        load_bfrags<4>(bf, ws + OFFW2 + nt * 16 * 128, 128, lane);
        int n = nt * 16 + rlo;
        float bias = ffb2[n];
        v8f d[3];
        gemm_tile_b3<4>(gbuf, SG, bf, lane, d);
#pragma unroll
        for (int mt = 0; mt < 3; ++mt)
#pragma unroll
            for (int v = 0; v < 8; ++v)
                cks[(mt * 16 + mo + v) * SCK + n] += d[mt][v] + bias;
    }
    __syncthreads();

    // ---- Head stage: flat[16][192] = concat(chunks[c]); hh1 = silu(flat @ hw1.T + hb1)
    for (int nt = wv; nt < 10; nt += NWAVES) {
        v8f acc = {};
#pragma unroll
        for (int ks = 0; ks < 6; ++ks) {
            int k0 = ks * 32;
            int c  = k0 / 64;           // which chunk supplies this K-range
            int fb = (k0 % 64) + lh8;
            const float* row = cks + (c * 16 + rlo) * SCK + fb;
            v4f c0 = *(const v4f*)(row);
            v4f c1 = *(const v4f*)(row + 4);
            v4f c2 = *(const v4f*)(row + 16);
            v4f c3 = *(const v4f*)(row + 20);
            v16h a;
#pragma unroll
            for (int i = 0; i < 4; ++i) {
                a[i]      = (half_t)c0[i];
                a[4 + i]  = (half_t)c1[i];
                a[8 + i]  = (half_t)c2[i];
                a[12 + i] = (half_t)c3[i];
            }
            v16h b = frag_ld(ws + OHW1 + (nt * 16 + rlo) * 192 + k0 + lh8);
            acc = wmma_f16(a, b, acc);
        }
        int n = nt * 16 + rlo;
        float bias = hb1[n];
#pragma unroll
        for (int v = 0; v < 8; ++v)
            hh1[(mo + v) * SHH + n] = (half_t)silu_f(acc[v] + bias);
    }
    __syncthreads();

    // ---- hh2 = silu(hh1 @ hw2.T + hb2) -> hh2f[16][64] f32 (overlays gbuf)
    for (int nt = wv; nt < 4; nt += NWAVES) {
        v16h bf[5];
        load_bfrags<5>(bf, ws + OHW2 + nt * 16 * 160, 160, lane);
        v8f d = gemm_tile_b<5>(hh1, SHH, bf, lane);
        int n = nt * 16 + rlo;
        float bias = hb2[n];
#pragma unroll
        for (int v = 0; v < 8; ++v)
            hh2f[(mo + v) * SH2 + n] = silu_f(d[v] + bias);
    }
    __syncthreads();

    // ---- Final: out[r] = hh2[r] . hw3 + hb3
    if (tid < 16) {
        const v4f* row4 = (const v4f*)(hh2f + tid * SH2);
        const v4f* w4   = (const v4f*)hw3;
        float acc = hb3[0];
#pragma unroll
        for (int i = 0; i < 16; ++i) {
            v4f a = row4[i], w = w4[i];
            acc += a[0] * w[0] + a[1] * w[1] + a[2] * w[2] + a[3] * w[3];
        }
        out[row0 + tid] = acc;
    }
}

// ---------------------------------------------------------------------------
extern "C" void kernel_launch(void* const* d_in, const int* in_sizes, int n_in,
                              void* d_out, int out_size, void* d_ws, size_t ws_size,
                              hipStream_t stream) {
    const float* x     = (const float*)d_in[0];
    const float* w_emb = (const float*)d_in[1];
    const float* pos_d = (const float*)d_in[2];
    const float* pos_c = (const float*)d_in[3];
    const float* tw1   = (const float*)d_in[4];
    const float* tb1   = (const float*)d_in[5];
    const float* tw2   = (const float*)d_in[6];
    const float* tb2   = (const float*)d_in[7];
    const float* tw3   = (const float*)d_in[8];
    const float* tb3   = (const float*)d_in[9];
    const float* ln1g  = (const float*)d_in[10];
    const float* ln1b  = (const float*)d_in[11];
    const float* inpw  = (const float*)d_in[12];
    const float* inpb  = (const float*)d_in[13];
    const float* outw  = (const float*)d_in[14];
    const float* outb  = (const float*)d_in[15];
    const float* ln2g  = (const float*)d_in[16];
    const float* ln2b  = (const float*)d_in[17];
    const float* ffw1  = (const float*)d_in[18];
    const float* ffb1  = (const float*)d_in[19];
    const float* ffw2  = (const float*)d_in[20];
    const float* ffb2  = (const float*)d_in[21];
    const float* hw1   = (const float*)d_in[22];
    const float* hb1   = (const float*)d_in[23];
    const float* hw2   = (const float*)d_in[24];
    const float* hb2   = (const float*)d_in[25];
    const float* hw3   = (const float*)d_in[26];
    const float* hb3   = (const float*)d_in[27];

    half_t* ws = (half_t*)d_ws;
    prep_kernel<<<64, 256, 0, stream>>>(w_emb, pos_d, tw1, tw2, tw3, inpw, outw,
                                        ffw1, ffw2, hw1, hw2, ws);

    int B = in_sizes[0] / 90;
    towers_main<<<B / 16, NTHREADS, 0, stream>>>(
        x, pos_c, tb1, tb2, tb3, ln1g, ln1b, inpb, outb, ln2g, ln2b,
        ffb1, ffb2, hb1, hb2, hw3, hb3, ws, (float*)d_out);
}
